// Conv_3d_net_4681514353326
// MI455X (gfx1250) — matrix operand order repaired and checked by reading, not compiled, not run on hardware
//
#include <hip/hip_runtime.h>
#include <cstddef>

// ---------------------------------------------------------------------------
// CDNA5 (gfx1250) implicit-GEMM 3D conv backbone using V_WMMA_F32_16X16X32_F16.
// Activations: f16 channels-last [D][H][W][C].  Weights: f16 [Cout][Kpad],
// K = tap*Cin + ci, zero-padded to Kpad = ceil(Cin*kvol/32)*32.
// One wave owns a 16-voxel M-tile and ALL Cout (NT = Cout/16 accumulators):
// each gathered A fragment feeds NT back-to-back v_wmma ops.  The epilogue
// transposes the C tile through a private LDS slab so every global write of
// intermediate activations is a coalesced b128 store.
// ---------------------------------------------------------------------------

typedef __attribute__((ext_vector_type(16))) _Float16 v16h;
typedef __attribute__((ext_vector_type(8)))  _Float16 v8h;
typedef __attribute__((ext_vector_type(8)))  float    v8f;

#define WAVES_PER_BLOCK 4

template<int CIN, int KD, int KH, int KW,
         int SD, int SH, int SW,
         int PD, int PH, int PW, int NT, bool FINAL>
__global__ __launch_bounds__(32 * WAVES_PER_BLOCK)
void conv_wmma(const _Float16* __restrict__ act_in,
               const _Float16* __restrict__ wf16,
               const float* __restrict__ scale,
               const float* __restrict__ shift,
               const float* __restrict__ maskv,
               _Float16* __restrict__ act_out,
               float* __restrict__ out32,
               int ID, int IH, int IW,
               int OD, int OH, int OW,
               int Kpad)
{
  constexpr int KREAL = CIN * KD * KH * KW;
  constexpr int COUT  = NT * 16;
  const int spatial = OD * OH * OW;
  const int lane = (int)(threadIdx.x & 31);
  const int wv   = (int)(threadIdx.x >> 5);
  const int nmt  = (spatial + 15) >> 4;
  const int mtile = blockIdx.x * WAVES_PER_BLOCK + wv;
  if (mtile >= nmt) return;                   // wave-uniform: EXEC stays all-1s

  const int row = lane & 15;
  const int hi  = lane >> 4;

  // ---- A row (M = output voxel) coordinates (32-bit math only) ----
  const int mg = mtile * 16 + row;
  const bool mv = mg < spatial;
  const unsigned om = mv ? (unsigned)mg : 0u;
  const int ow = (int)(om % (unsigned)OW);
  const unsigned t0 = om / (unsigned)OW;
  const int oh = (int)(t0 % (unsigned)OH);
  const int od = (int)(t0 / (unsigned)OH);
  const int bd = od * SD - PD;
  const int bh = oh * SH - PH;
  const int bw = ow * SW - PW;

  // ---- B base: lane = one N column (co = nt*16 + row), K = hi*8+{0..7} and 16+hi*8+{0..7} ----
  const _Float16* __restrict__ wrow0 = wf16 + (size_t)row * Kpad + hi * 8;

  v8f acc[NT] = {};
  for (int k0 = 0; k0 < Kpad; k0 += 32) {
    // ---- A fragment: lane holds K = hi*8+{0..7} and 16+hi*8+{0..7} ----
    v16h a = {};
    if constexpr (CIN >= 8) {
      // each 8-run is contiguous channels-last (never crosses a tap)
#pragma unroll
      for (int hseg = 0; hseg < 2; ++hseg) {
        const int kk0 = k0 + hseg * 16 + hi * 8;
        v8h av = {};
        if (mv && kk0 < KREAL) {
          const int ci0 = kk0 % CIN;            // compile-time masks/shifts
          const int tap = kk0 / CIN;
          const int kw = tap % KW;
          const int t1 = tap / KW;
          const int kh = t1 % KH;
          const int kd = t1 / KH;
          const int id = bd + kd, ih = bh + kh, iw = bw + kw;
          if ((unsigned)id < (unsigned)ID && (unsigned)ih < (unsigned)IH &&
              (unsigned)iw < (unsigned)IW)
            av = *(const v8h*)(act_in +
                 (size_t)((id * IH + ih) * IW + iw) * CIN + ci0);
        }
#pragma unroll
        for (int e = 0; e < 8; ++e) a[hseg * 8 + e] = av[e];
      }
    } else {
#pragma unroll
      for (int e = 0; e < 16; ++e) {
        const int kk = k0 + hi * 8 + (e < 8 ? e : e + 8);
        _Float16 val = (_Float16)0.f;
        if (mv && kk < KREAL) {
          const int ci  = kk % CIN;
          const int tap = kk / CIN;
          const int kw = tap % KW;
          const int t1 = tap / KW;
          const int kh = t1 % KH;
          const int kd = t1 / KH;
          const int id = bd + kd, ih = bh + kh, iw = bw + kw;
          if ((unsigned)id < (unsigned)ID && (unsigned)ih < (unsigned)IH &&
              (unsigned)iw < (unsigned)IW)
            val = act_in[(size_t)((id * IH + ih) * IW + iw) * CIN + ci];
        }
        a[e] = val;
      }
    }

    // ---- NT x (B load + WMMA), reusing the A fragment ----
#pragma unroll
    for (int nt = 0; nt < NT; ++nt) {
      const _Float16* __restrict__ w = wrow0 + (size_t)nt * 16 * Kpad + k0;
      __builtin_prefetch(w + 128, 0, 1);        // global_prefetch_b8 ahead
      v16h b;
      const v8h b0 = *(const v8h*)(w);
      const v8h b1 = *(const v8h*)(w + 16);
#pragma unroll
      for (int e = 0; e < 8; ++e) { b[e] = b0[e]; b[8 + e] = b1[e]; }
      acc[nt] = __builtin_amdgcn_wmma_f32_16x16x32_f16(
          false, a, false, b, (short)0, acc[nt], false, false);
    }
  }

  // ---- epilogue: BN affine + ReLU + mask; C layout M = r + 8*hi, N = row ----
  float sc[NT], sh[NT];
#pragma unroll
  for (int nt = 0; nt < NT; ++nt) {
    sc[nt] = scale[nt * 16 + row];
    sh[nt] = shift[nt * 16 + row];
  }

  if constexpr (FINAL) {
#pragma unroll
    for (int r = 0; r < 8; ++r) {
      const int mm = mtile * 16 + hi * 8 + r;
      if (mm < spatial) {
        const float mk = maskv[mm];
#pragma unroll
        for (int nt = 0; nt < NT; ++nt) {
          const int co = nt * 16 + row;
          float v = acc[nt][r] * sc[nt] + sh[nt];
          v = v > 0.f ? v : 0.f;
          out32[(size_t)co * spatial + mm] = v * mk;        // NCDHW flat
        }
      }
    }
  } else {
    // per-wave private LDS slab: scatter f16 tile, read back linearly,
    // emit coalesced b128 global stores (LDS ops are in-order per wave).
    __shared__ _Float16 stile[WAVES_PER_BLOCK][16 * 128];
    _Float16* __restrict__ lbase = &stile[wv][0];
#pragma unroll
    for (int r = 0; r < 8; ++r) {
      const int mrow = hi * 8 + r;
      const int mm = mtile * 16 + mrow;
      const float mk = (mm < spatial) ? maskv[mm] : 0.f;
#pragma unroll
      for (int nt = 0; nt < NT; ++nt) {
        float v = acc[nt][r] * sc[nt] + sh[nt];
        v = v > 0.f ? v : 0.f;
        lbase[mrow * COUT + nt * 16 + row] = (_Float16)(v * mk);
      }
    }
    constexpr int ROWB  = COUT * 2;             // bytes per voxel row
    constexpr int TILEB = 16 * ROWB;            // tile bytes
    constexpr int PASSES = TILEB / 512;         // 32 lanes x 16B per pass
#pragma unroll
    for (int p = 0; p < PASSES; ++p) {
      const int boff = p * 512 + lane * 16;
      const int mrow = boff / ROWB;             // compile-time divisor
      const int coff = boff % ROWB;
      const int mm = mtile * 16 + mrow;
      if (mm < spatial) {
        const v8h v = *(const v8h*)((const char*)lbase + boff);
        *(v8h*)((char*)act_out + (size_t)mm * ROWB + coff) = v;
      }
    }
  }
}

// ---- mask max-pool (reduce_window max > 0) ----
__global__ void maskpool(const float* __restrict__ mi, float* __restrict__ mo,
                         int ID, int IH, int IW, int OD, int OH, int OW,
                         int KDk, int KHk, int KWk,
                         int SDk, int SHk, int SWk,
                         int PDk, int PHk, int PWk)
{
  const int m = blockIdx.x * blockDim.x + threadIdx.x;
  const int sp = OD * OH * OW;
  if (m >= sp) return;
  const int ow = (int)((unsigned)m % (unsigned)OW);
  const unsigned t = (unsigned)m / (unsigned)OW;
  const int oh = (int)(t % (unsigned)OH);
  const int od = (int)(t / (unsigned)OH);
  float v = 0.f;
  for (int kd = 0; kd < KDk; ++kd) {
    const int id = od * SDk - PDk + kd;
    if ((unsigned)id >= (unsigned)ID) continue;
    for (int kh = 0; kh < KHk; ++kh) {
      const int ih = oh * SHk - PHk + kh;
      if ((unsigned)ih >= (unsigned)IH) continue;
      for (int kw = 0; kw < KWk; ++kw) {
        const int iw = ow * SWk - PWk + kw;
        if ((unsigned)iw >= (unsigned)IW) continue;
        const float x = mi[(size_t)(id * IH + ih) * IW + iw];
        v = v > x ? v : x;
      }
    }
  }
  mo[m] = v > 0.f ? 1.f : 0.f;
}

// ---- weight repack: f32 [O][Ci][kd][kh][kw] -> f16 [O][Kpad], K=tap*Ci+ci ----
__global__ void wconv(const float* __restrict__ w, _Float16* __restrict__ wf,
                      int Cout, int CIN, int T, int Kpad)
{
  const int idx = blockIdx.x * blockDim.x + threadIdx.x;
  const int total = Cout * Kpad;
  if (idx >= total) return;
  const int o  = (int)((unsigned)idx / (unsigned)Kpad);
  const int kk = (int)((unsigned)idx % (unsigned)Kpad);
  float v = 0.f;
  if (kk < CIN * T) {
    const int ci  = (int)((unsigned)kk % (unsigned)CIN);
    const int tap = (int)((unsigned)kk / (unsigned)CIN);
    v = w[(size_t)o * CIN * T + (size_t)ci * T + tap];
  }
  wf[idx] = (_Float16)v;
}

// ---- fold BN into scale/shift ----
__global__ void bnprep(const float* __restrict__ bn,
                       float* __restrict__ sc, float* __restrict__ sh, int C)
{
  const int c = blockIdx.x * blockDim.x + threadIdx.x;
  if (c >= C) return;
  const float g = bn[c], b = bn[C + c], m = bn[2 * C + c], v = bn[3 * C + c];
  const float s = g * rsqrtf(v + 1e-3f);
  sc[c] = s;
  sh[c] = b - m * s;
}

// ---- input f32 NCDHW -> f16 channels-last ----
__global__ void xconv(const float* __restrict__ x, _Float16* __restrict__ a,
                      int S, int C)
{
  const int i = blockIdx.x * blockDim.x + threadIdx.x;
  if (i >= S * C) return;
  const int m = (int)((unsigned)i / (unsigned)C);
  const int c = (int)((unsigned)i % (unsigned)C);
  a[(size_t)m * C + c] = (_Float16)x[(size_t)c * S + m];
}

extern "C" void kernel_launch(void* const* d_in, const int* in_sizes, int n_in,
                              void* d_out, int out_size, void* d_ws, size_t ws_size,
                              hipStream_t stream)
{
  (void)in_sizes; (void)n_in; (void)out_size; (void)ws_size;
  const float* x     = (const float*)d_in[0];
  const float* mask0 = (const float*)d_in[1];
  const float* Wt[12]; const float* BN[12];
  for (int i = 0; i < 12; ++i) { Wt[i] = (const float*)d_in[2 + 2 * i];
                                 BN[i] = (const float*)d_in[3 + 2 * i]; }

  const int D0 = 41, H0 = 200, W0 = 176; const long S0 = (long)D0 * H0 * W0;
  const int D2 = 21, H2 = 100, W2 = 88;  const long S2 = (long)D2 * H2 * W2;
  const int D3 = 11, H3 = 50,  W3 = 44;  const long S3 = (long)D3 * H3 * W3;
  const int D4 = 5,  H4 = 25,  W4 = 22;  const long S4 = (long)D4 * H4 * W4;
  const int D5 = 2,  H5 = 25,  W5 = 22;  const long S5 = (long)D5 * H5 * W5;

  static const int CinA[12]  = {4,16,16,32,32,32,64,64,64,64,64,64};
  static const int CoutA[12] = {16,16,32,32,32,64,64,64,64,64,64,128};
  static const int TA[12]    = {27,27,27,27,27,27,27,27,27,27,27,3};

  size_t off = 0;
  auto alloc = [&](size_t bytes) -> size_t {
    size_t o = off; off += (bytes + 255) & ~(size_t)255; return o;
  };
  char* ws = (char*)d_ws;
  const size_t o_act0 = alloc((size_t)S0 * 4  * sizeof(_Float16));
  const size_t o_bufA = alloc((size_t)S0 * 16 * sizeof(_Float16));
  const size_t o_bufB = alloc((size_t)S0 * 16 * sizeof(_Float16));
  const size_t o_m2 = alloc(S2 * 4), o_m3 = alloc(S3 * 4);
  const size_t o_m4 = alloc(S4 * 4), o_m5 = alloc(S5 * 4);
  size_t o_wf[12], o_sc[12], o_sh[12]; int Kp[12];
  for (int i = 0; i < 12; ++i) {
    Kp[i] = ((CinA[i] * TA[i] + 31) / 32) * 32;
    o_wf[i] = alloc((size_t)CoutA[i] * Kp[i] * sizeof(_Float16));
    o_sc[i] = alloc(CoutA[i] * 4);
    o_sh[i] = alloc(CoutA[i] * 4);
  }

  // ---- prep: weights, BN fold, input convert, masks ----
  for (int i = 0; i < 12; ++i) {
    const int tot = CoutA[i] * Kp[i];
    wconv<<<(tot + 255) / 256, 256, 0, stream>>>(
        Wt[i], (_Float16*)(ws + o_wf[i]), CoutA[i], CinA[i], TA[i], Kp[i]);
    bnprep<<<(CoutA[i] + 255) / 256, 256, 0, stream>>>(
        BN[i], (float*)(ws + o_sc[i]), (float*)(ws + o_sh[i]), CoutA[i]);
  }
  xconv<<<(int)((S0 * 4 + 255) / 256), 256, 0, stream>>>(
      x, (_Float16*)(ws + o_act0), (int)S0, 4);
  maskpool<<<(int)((S2 + 255) / 256), 256, 0, stream>>>(
      mask0, (float*)(ws + o_m2), D0,H0,W0, D2,H2,W2, 3,3,3, 2,2,2, 1,1,1);
  maskpool<<<(int)((S3 + 255) / 256), 256, 0, stream>>>(
      (float*)(ws + o_m2), (float*)(ws + o_m3), D2,H2,W2, D3,H3,W3, 3,3,3, 2,2,2, 1,1,1);
  maskpool<<<(int)((S4 + 255) / 256), 256, 0, stream>>>(
      (float*)(ws + o_m3), (float*)(ws + o_m4), D3,H3,W3, D4,H4,W4, 3,3,3, 2,2,2, 0,1,1);
  maskpool<<<(int)((S5 + 255) / 256), 256, 0, stream>>>(
      (float*)(ws + o_m4), (float*)(ws + o_m5), D4,H4,W4, D5,H5,W5, 3,1,1, 2,1,1, 0,0,0);

  auto grid = [](long spatial) -> int {
    const long nmt = (spatial + 15) / 16;
    return (int)((nmt + WAVES_PER_BLOCK - 1) / WAVES_PER_BLOCK);
  };
  const _Float16* A = (const _Float16*)(ws + o_bufA);
  const _Float16* B = (const _Float16*)(ws + o_bufB);
  _Float16* Aw = (_Float16*)(ws + o_bufA);
  _Float16* Bw = (_Float16*)(ws + o_bufB);
#define SC(i) (const float*)(ws + o_sc[i]), (const float*)(ws + o_sh[i])
#define WF(i) (const _Float16*)(ws + o_wf[i])
#define MK(o) (const float*)(ws + (o))

  // L0: subm 4->16 @ full res           act0 -> A
  conv_wmma<4,3,3,3, 1,1,1, 1,1,1, 1, false><<<grid(S0), 128, 0, stream>>>(
      (const _Float16*)(ws + o_act0), WF(0), SC(0), mask0, Aw, nullptr,
      D0,H0,W0, D0,H0,W0, Kp[0]);
  // L1: subm 16->16                     A -> B
  conv_wmma<16,3,3,3, 1,1,1, 1,1,1, 1, false><<<grid(S0), 128, 0, stream>>>(
      A, WF(1), SC(1), mask0, Bw, nullptr, D0,H0,W0, D0,H0,W0, Kp[1]);
  // L2: spconv 16->32 s2 p1             B -> A
  conv_wmma<16,3,3,3, 2,2,2, 1,1,1, 2, false><<<grid(S2), 128, 0, stream>>>(
      B, WF(2), SC(2), MK(o_m2), Aw, nullptr, D0,H0,W0, D2,H2,W2, Kp[2]);
  // L3: subm 32->32                     A -> B
  conv_wmma<32,3,3,3, 1,1,1, 1,1,1, 2, false><<<grid(S2), 128, 0, stream>>>(
      A, WF(3), SC(3), MK(o_m2), Bw, nullptr, D2,H2,W2, D2,H2,W2, Kp[3]);
  // L4: subm 32->32                     B -> A
  conv_wmma<32,3,3,3, 1,1,1, 1,1,1, 2, false><<<grid(S2), 128, 0, stream>>>(
      B, WF(4), SC(4), MK(o_m2), Aw, nullptr, D2,H2,W2, D2,H2,W2, Kp[4]);
  // L5: spconv 32->64 s2 p1             A -> B
  conv_wmma<32,3,3,3, 2,2,2, 1,1,1, 4, false><<<grid(S3), 128, 0, stream>>>(
      A, WF(5), SC(5), MK(o_m3), Bw, nullptr, D2,H2,W2, D3,H3,W3, Kp[5]);
  // L6: subm 64->64                     B -> A
  conv_wmma<64,3,3,3, 1,1,1, 1,1,1, 4, false><<<grid(S3), 128, 0, stream>>>(
      B, WF(6), SC(6), MK(o_m3), Aw, nullptr, D3,H3,W3, D3,H3,W3, Kp[6]);
  // L7: subm 64->64                     A -> B
  conv_wmma<64,3,3,3, 1,1,1, 1,1,1, 4, false><<<grid(S3), 128, 0, stream>>>(
      A, WF(7), SC(7), MK(o_m3), Bw, nullptr, D3,H3,W3, D3,H3,W3, Kp[7]);
  // L8: spconv 64->64 s2 p(0,1,1)       B -> A
  conv_wmma<64,3,3,3, 2,2,2, 0,1,1, 4, false><<<grid(S4), 128, 0, stream>>>(
      B, WF(8), SC(8), MK(o_m4), Aw, nullptr, D3,H3,W3, D4,H4,W4, Kp[8]);
  // L9: subm 64->64                     A -> B
  conv_wmma<64,3,3,3, 1,1,1, 1,1,1, 4, false><<<grid(S4), 128, 0, stream>>>(
      A, WF(9), SC(9), MK(o_m4), Bw, nullptr, D4,H4,W4, D4,H4,W4, Kp[9]);
  // L10: subm 64->64                    B -> A
  conv_wmma<64,3,3,3, 1,1,1, 1,1,1, 4, false><<<grid(S4), 128, 0, stream>>>(
      B, WF(10), SC(10), MK(o_m4), Aw, nullptr, D4,H4,W4, D4,H4,W4, Kp[10]);
  // L11: spconv 64->128 k(3,1,1) s(2,1,1) p0 -> d_out (f32 NCDHW == reshape)
  conv_wmma<64,3,1,1, 2,1,1, 0,0,0, 8, true><<<grid(S5), 128, 0, stream>>>(
      A, WF(11), SC(11), MK(o_m5), nullptr, (float*)d_out,
      D4,H4,W4, D5,H5,W5, Kp[11]);

#undef SC
#undef WF
#undef MK
}
